// AttentionBlock_13383118094897
// MI455X (gfx1250) — compile-verified, matrix operand order checked by reading, not run on hardware
//
#include <hip/hip_runtime.h>
#include <hip/hip_bf16.h>
#include <math.h>
#include <stdint.h>

// ---------------------------------------------------------------------------
// AttentionBlock (GroupNorm -> QKV -> MHSA -> Proj -> residual) for MI455X
// (gfx1250, wave32). GEMMs: v_wmma_f32_16x16x32_f16 with the B panel staged
// in LDS via double-buffered GLOBAL_LOAD_ASYNC_TO_LDS_B128 (ASYNCcnt).
// Softmax / GroupNorm statistics stay in f32.
//
// Shapes: B=8, C=512, H=W=32 -> N=1024, heads=8, d=64, groups=32.
// Workspace requirement: ~44.1 MB (see layout in kernel_launch).
// ---------------------------------------------------------------------------

typedef __attribute__((ext_vector_type(16))) _Float16 v16h;
typedef __attribute__((ext_vector_type(8)))  _Float16 v8h;
typedef __attribute__((ext_vector_type(8)))  float    v8f;

#define BB   8
#define CC   512
#define NN   1024
#define NH   8
#define DD   64
#define NG   32

// ---------------------------------------------------------------------------
// Fragment loader for the CDNA5 16-bit 16x32 A (and, by symmetry, 32x16 B)
// layout. Memory is row-major with `ld` halves per row; `row` is the M (or N)
// index for this lane; k0 is the K-tile base. Lane halves:
//   lanes 0-15 : K = k0 + {0..7, 16..23};  lanes 16-31: K = k0 + {8..15, 24..31}
// -> two 16-byte contiguous loads per lane (global_load_b128 / ds_load_b128).
// ---------------------------------------------------------------------------
__device__ __forceinline__ v16h load_frag(const _Float16* base, int ld,
                                          int row, int k0, int hi) {
  const _Float16* p = base + (size_t)row * (size_t)ld + k0 + (hi << 3);
  const v8h lo = *reinterpret_cast<const v8h*>(p);
  const v8h up = *reinterpret_cast<const v8h*>(p + 16);
  v16h f;
#pragma unroll
  for (int i = 0; i < 8; ++i) { f[i] = lo[i]; f[i + 8] = up[i]; }
  return f;
}

__device__ __forceinline__ v8f wmma16(v16h a, v16h b, v8f c) {
  return __builtin_amdgcn_wmma_f32_16x16x32_f16(false, a, false, b,
                                                (short)0, c, false, false);
}

// ---------------------------------------------------------------------------
// Async global->LDS copy (CDNA5): 16 bytes per lane, tracked by ASYNCcnt.
// lds_off is the raw LDS byte offset (low 32 bits of the generic pointer).
// ---------------------------------------------------------------------------
__device__ __forceinline__ void async_b128(uint32_t lds_off, const _Float16* g) {
  asm volatile("global_load_async_to_lds_b128 %0, %1, off"
               :: "v"(lds_off), "v"((unsigned long long)(size_t)g)
               : "memory");
}
__device__ __forceinline__ void wait_async_le2() {
  asm volatile("s_wait_asynccnt 2" ::: "memory");
}
__device__ __forceinline__ void wait_async_0() {
  asm volatile("s_wait_asynccnt 0" ::: "memory");
}
__device__ __forceinline__ uint32_t lds_off32(const void* p) {
  return (uint32_t)(unsigned long long)(size_t)p;
}

__device__ __forceinline__ float redmax16(float v) {
  v = fmaxf(v, __shfl_xor(v, 1, 16));
  v = fmaxf(v, __shfl_xor(v, 2, 16));
  v = fmaxf(v, __shfl_xor(v, 4, 16));
  v = fmaxf(v, __shfl_xor(v, 8, 16));
  return v;
}
__device__ __forceinline__ float redsum16(float v) {
  v += __shfl_xor(v, 1, 16);
  v += __shfl_xor(v, 2, 16);
  v += __shfl_xor(v, 4, 16);
  v += __shfl_xor(v, 8, 16);
  return v;
}

// Stage one 64(N) x 32(K) f16 B panel (4KB) into LDS: 256 16B chunks,
// 2 per thread (128-thread block). Rows are `Bpanel` rows n0.. (ld=512).
__device__ __forceinline__ void stage_B(const _Float16* Bpanel, int k0,
                                        uint32_t lds_base, int tid) {
#pragma unroll
  for (int j = 0; j < 2; ++j) {
    int c    = tid + j * 128;      // 0..255
    int row  = c >> 2;             // 0..63
    int col8 = (c & 3) * 8;        // 0,8,16,24 halves
    async_b128(lds_base + (uint32_t)(row * 64 + col8 * 2),
               Bpanel + (size_t)row * CC + k0 + col8);
  }
}

// ---------------------------------------------------------------------------
// Kernel 0: cast qkv_w [1536*512] and proj_w [512*512] f32 -> f16.
// ---------------------------------------------------------------------------
__global__ void cast_weights(const float* __restrict__ qkv_w,
                             const float* __restrict__ proj_w,
                             _Float16* __restrict__ wq_h,
                             _Float16* __restrict__ wp_h) {
  int i = blockIdx.x * 256 + threadIdx.x;
  const int n1 = 3 * CC * CC;          // 786432
  if (i < n1)            wq_h[i]      = (_Float16)qkv_w[i];
  else                   wp_h[i - n1] = (_Float16)proj_w[i - n1];
}

// ---------------------------------------------------------------------------
// Kernel 1: GroupNorm statistics. One block per (b, g); each group is a
// contiguous run of 16*1024 floats in x[b][c][n]. stats[bg] = {mean, rstd}.
// ---------------------------------------------------------------------------
__global__ void gn_stats(const float* __restrict__ x, float* __restrict__ stats) {
  __shared__ float ss[256], sq[256];
  const int bg = blockIdx.x;                 // 0..255
  const int tid = threadIdx.x;
  const float* p = x + (size_t)bg * 16384;
  float s = 0.f, q = 0.f;
  for (int i = tid; i < 16384; i += 256) {
    float v = p[i]; s += v; q += v * v;
  }
  ss[tid] = s; sq[tid] = q;
  __syncthreads();
  for (int st = 128; st > 0; st >>= 1) {
    if (tid < st) { ss[tid] += ss[tid + st]; sq[tid] += sq[tid + st]; }
    __syncthreads();
  }
  if (tid == 0) {
    float mean = ss[0] * (1.f / 16384.f);
    float var  = sq[0] * (1.f / 16384.f) - mean * mean;
    stats[2 * bg]     = mean;
    stats[2 * bg + 1] = rsqrtf(var + 1e-6f);
  }
}

// ---------------------------------------------------------------------------
// Kernel 2: apply GroupNorm + affine, write transposed f16 xn_t[b][n][c]
// (K-contiguous B-operand for the QKV GEMM).
// ---------------------------------------------------------------------------
__global__ void gn_apply(const float* __restrict__ x,
                         const float* __restrict__ stats,
                         const float* __restrict__ nw,
                         const float* __restrict__ nb,
                         _Float16* __restrict__ xn_t) {
  int i = blockIdx.x * 256 + threadIdx.x;      // 0 .. B*C*N-1
  int b   = i >> 19;
  int rem = i & 524287;
  int n   = rem >> 9;
  int c   = rem & 511;
  int g   = c >> 4;
  float mean = stats[(b * NG + g) * 2];
  float rstd = stats[(b * NG + g) * 2 + 1];
  float v = x[((size_t)b * CC + c) * NN + n];
  v = (v - mean) * rstd * nw[c] + nb[c];
  xn_t[(size_t)b * (CC * NN) + (size_t)n * CC + c] = (_Float16)v;
}

// ---------------------------------------------------------------------------
// Shared GEMM core: block = 4 waves, block tile 64(M) x 64(N), K=512.
// B panel (64x32 per k-step) staged in LDS via double-buffered async copies
// and shared by all 4 waves; each wave keeps its private A row-strip from
// global (no cross-wave A reuse exists). acc[4] = four 16x16 f32 C tiles.
// ---------------------------------------------------------------------------
__device__ __forceinline__ void gemm_core(const _Float16* __restrict__ A,
                                          int m0,                 // wave's M base
                                          const _Float16* __restrict__ Bpanel, // row n0
                                          _Float16* lds0, _Float16* lds1,
                                          int tid, int hi, int lr,
                                          v8f acc[4]) {
#pragma unroll
  for (int t = 0; t < 4; ++t)
#pragma unroll
    for (int r = 0; r < 8; ++r) acc[t][r] = 0.f;

  const uint32_t l0 = lds_off32(lds0);
  const uint32_t l1 = lds_off32(lds1);

  stage_B(Bpanel, 0, l0, tid);                       // prologue: buf0 <- k=0
  for (int ks = 0; ks < 16; ++ks) {
    const int k0 = ks * 32;
    const _Float16* cur = (ks & 1) ? lds1 : lds0;
    if (ks < 15) {
      stage_B(Bpanel, k0 + 32, (ks & 1) ? l0 : l1, tid);
      wait_async_le2();    // own 2 chunks of step ks complete (in-order)
    } else {
      wait_async_0();
    }
    __syncthreads();       // all waves' chunks visible

    v16h a = load_frag(A, CC, m0 + lr, k0, hi);      // global A fragment
#pragma unroll
    for (int t = 0; t < 4; ++t) {
      v16h bf = load_frag(cur, 32, t * 16 + lr, 0, hi);  // ds_load_b128 x2
      acc[t] = wmma16(a, bf, acc[t]);
    }
    __syncthreads();       // consumption done before buffer reuse
  }
}

// ---------------------------------------------------------------------------
// Kernel 3: QKV GEMM. Y[o,n] = sum_c W[o,c] * xn[c,n] + bias[o], per batch.
// Grid: 8 batches x 24 (M/64) x 16 (N/64) blocks of 128 threads. Epilogue:
//   o <  512 : q_t[b][h][n][d] = 0.125 * y   (scale folded into Q)
//   o < 1024 : k_t[b][h][n][d] = y
//   else     : v_b[b][h][d][n] = y
// ---------------------------------------------------------------------------
__global__ void qkv_gemm(const _Float16* __restrict__ W,
                         const _Float16* __restrict__ X,
                         const float* __restrict__ bias,
                         _Float16* __restrict__ q_t,
                         _Float16* __restrict__ k_t,
                         _Float16* __restrict__ v_b) {
  __shared__ _Float16 bstage[2][64 * 32];            // 2 x 4KB
  const int tid  = threadIdx.x;
  const int lane = tid & 31;
  const int wv   = tid >> 5;
  const int hi   = lane >> 4;
  const int lr   = lane & 15;
  int blk = blockIdx.x;                              // 0..3071
  int b   = blk / 384;
  int rem = blk - b * 384;
  int mt64 = rem >> 4;                               // 0..23
  int ns   = rem & 15;                               // 0..15
  int m0   = mt64 * 64 + wv * 16;
  int n0   = ns * 64;
  const _Float16* Xb = X + (size_t)b * (CC * NN) + (size_t)n0 * CC;

  v8f acc[4];
  gemm_core(W, m0, Xb, bstage[0], bstage[1], tid, hi, lr, acc);

  const int cls = m0 >> 9;                    // 0=q 1=k 2=v (tile never straddles)
  const int h   = (m0 & 511) >> 6;            // constant over the 16-row tile
#pragma unroll
  for (int t = 0; t < 4; ++t) {
    int n = n0 + t * 16 + lr;
#pragma unroll
    for (int r = 0; r < 8; ++r) {
      int o = m0 + r + 8 * hi;
      float val = acc[t][r] + bias[o];
      int dd = o & 63;
      if (cls == 0)
        q_t[(((size_t)b * NH + h) * NN + n) * DD + dd] = (_Float16)(val * 0.125f);
      else if (cls == 1)
        k_t[(((size_t)b * NH + h) * NN + n) * DD + dd] = (_Float16)val;
      else
        v_b[(((size_t)b * NH + h) * DD + dd) * NN + n] = (_Float16)val;
    }
  }
}

// ---------------------------------------------------------------------------
// Kernel 4: flash attention. One wave owns 16 query rows of one (b,h);
// iterates keys in tiles of 32 with online softmax (f32), P re-laid-out
// through LDS into the A-fragment layout, PV accumulated in 4 f32 C tiles.
// Output written transposed: ao_t[b][n][h*64+d] (K-contiguous for proj GEMM).
// ---------------------------------------------------------------------------
__global__ void attn_flash(const _Float16* __restrict__ q_t,
                           const _Float16* __restrict__ k_t,
                           const _Float16* __restrict__ v_b,
                           _Float16* __restrict__ ao_t) {
  __shared__ _Float16 pl[4][16 * 32];          // 1KB per wave
  const int lane = threadIdx.x & 31;
  const int wv   = threadIdx.x >> 5;
  const int hi   = lane >> 4;
  const int lr   = lane & 15;
  int gw = blockIdx.x * 4 + wv;                // 0..4095
  int bh = gw >> 6;                            // 0..63
  int nt = gw & 63;
  int b  = bh >> 3, h = bh & 7;
  int n0 = nt * 16;

  const _Float16* qh = q_t + (size_t)bh * NN * DD;
  const _Float16* kh = k_t + (size_t)bh * NN * DD;
  const _Float16* vh = v_b + (size_t)bh * DD * NN;
  _Float16* myp = pl[wv];

  // Q^T tile: A[M=16 rows n, K=64 d] as two 16x32 fragments (scale pre-folded).
  v16h a0 = load_frag(qh, DD, n0 + lr, 0, hi);
  v16h a1 = load_frag(qh, DD, n0 + lr, 32, hi);

  float mrun[8], lrun[8];
  v8f o[4];
#pragma unroll
  for (int r = 0; r < 8; ++r) { mrun[r] = -3.0e38f; lrun[r] = 0.f; }
#pragma unroll
  for (int t = 0; t < 4; ++t)
#pragma unroll
    for (int r = 0; r < 8; ++r) o[t][r] = 0.f;

  for (int m0 = 0; m0 < NN; m0 += 32) {
    // S tiles: s0 -> keys m0..m0+15, s1 -> m0+16..m0+31  (K over d=64)
    v8f s0, s1;
#pragma unroll
    for (int r = 0; r < 8; ++r) { s0[r] = 0.f; s1[r] = 0.f; }
    s0 = wmma16(a0, load_frag(kh, DD, m0 + lr, 0, hi), s0);
    s0 = wmma16(a1, load_frag(kh, DD, m0 + lr, 32, hi), s0);
    s1 = wmma16(a0, load_frag(kh, DD, m0 + 16 + lr, 0, hi), s1);
    s1 = wmma16(a1, load_frag(kh, DD, m0 + 16 + lr, 32, hi), s1);

    // Online softmax per row (row M=r+8*hi lives across 16 lanes of reg r).
#pragma unroll
    for (int r = 0; r < 8; ++r) {
      float mx = redmax16(fmaxf(s0[r], s1[r]));
      float mn = fmaxf(mrun[r], mx);
      float al = __expf(mrun[r] - mn);
      float e0 = __expf(s0[r] - mn);
      float e1 = __expf(s1[r] - mn);
      float rs = redsum16(e0 + e1);
      lrun[r] = lrun[r] * al + rs;
      mrun[r] = mn;
#pragma unroll
      for (int t = 0; t < 4; ++t) o[t][r] *= al;
      // P tile (16x32 f16) staged to LDS in (row, key) order.
      myp[(r + 8 * hi) * 32 + lr]      = (_Float16)e0;
      myp[(r + 8 * hi) * 32 + 16 + lr] = (_Float16)e1;
    }
    asm volatile("s_wait_dscnt 0" ::: "memory");   // wave-private LDS: stores visible

    // P as A fragment, V tiles as B fragments: O[n, d] += P * V^T
    v16h pa = load_frag(myp, 32, lr, 0, hi);
#pragma unroll
    for (int dt = 0; dt < 4; ++dt) {
      v16h vb = load_frag(vh, NN, dt * 16 + lr, m0, hi);
      o[dt] = wmma16(pa, vb, o[dt]);
    }
  }

  // Normalize and store transposed: ao_t[b][n][h*64 + d]
  _Float16* aob = ao_t + (size_t)b * (CC * NN);
#pragma unroll
  for (int dt = 0; dt < 4; ++dt) {
#pragma unroll
    for (int r = 0; r < 8; ++r) {
      int n = n0 + r + 8 * hi;
      int c = h * DD + dt * 16 + lr;
      aob[(size_t)n * CC + c] = (_Float16)(o[dt][r] / lrun[r]);
    }
  }
}

// ---------------------------------------------------------------------------
// Kernel 5: proj GEMM + bias + residual, f32 output.
// out[b,c,n] = sum_k Wp[c,k] * ao[b,n,k] + pb[c] + x[b,c,n]
// Grid: 8 batches x 8 (M/64) x 16 (N/64) blocks of 128 threads.
// ---------------------------------------------------------------------------
__global__ void proj_gemm(const _Float16* __restrict__ Wp,
                          const _Float16* __restrict__ A,
                          const float* __restrict__ pb,
                          const float* __restrict__ x,
                          float* __restrict__ out) {
  __shared__ _Float16 bstage[2][64 * 32];            // 2 x 4KB
  const int tid  = threadIdx.x;
  const int lane = tid & 31;
  const int wv   = tid >> 5;
  const int hi   = lane >> 4;
  const int lr   = lane & 15;
  int blk = blockIdx.x;                              // 0..1023
  int b   = blk >> 7;
  int rem = blk & 127;
  int mt64 = rem >> 4;                               // 0..7
  int ns   = rem & 15;
  int m0   = mt64 * 64 + wv * 16;
  int n0   = ns * 64;
  const _Float16* Ab = A + (size_t)b * (CC * NN) + (size_t)n0 * CC;

  v8f acc[4];
  gemm_core(Wp, m0, Ab, bstage[0], bstage[1], tid, hi, lr, acc);

#pragma unroll
  for (int t = 0; t < 4; ++t) {
    int n = n0 + t * 16 + lr;
#pragma unroll
    for (int r = 0; r < 8; ++r) {
      int c = m0 + r + 8 * hi;
      size_t idx = ((size_t)b * CC + c) * NN + n;
      out[idx] = acc[t][r] + pb[c] + x[idx];
    }
  }
}

// ---------------------------------------------------------------------------
// Host-side launcher (graph-capture safe: only kernel launches on `stream`).
// ---------------------------------------------------------------------------
extern "C" void kernel_launch(void* const* d_in, const int* in_sizes, int n_in,
                              void* d_out, int out_size, void* d_ws, size_t ws_size,
                              hipStream_t stream) {
  const float* x      = (const float*)d_in[0];
  const float* norm_w = (const float*)d_in[1];
  const float* norm_b = (const float*)d_in[2];
  const float* qkv_w  = (const float*)d_in[3];
  const float* qkv_b  = (const float*)d_in[4];
  const float* proj_w = (const float*)d_in[5];
  const float* proj_b = (const float*)d_in[6];
  float* out = (float*)d_out;

  // Workspace layout (needs ~44.1 MB):
  //   [0, 2048)            f32 stats  (256 groups x {mean, rstd})
  //   then f16 arena:
  float*    stats = (float*)d_ws;
  _Float16* hb    = (_Float16*)((char*)d_ws + 2048);
  _Float16* wq_h  = hb;                               //  786432 halves
  _Float16* wp_h  = wq_h + (size_t)3 * CC * CC;       //  262144
  _Float16* xn_t  = wp_h + (size_t)CC * CC;           // 4194304  [b][n][c]
  _Float16* q_t   = xn_t + (size_t)BB * CC * NN;      // 4194304  [b][h][n][d]
  _Float16* k_t   = q_t  + (size_t)BB * CC * NN;      // 4194304  [b][h][n][d]
  _Float16* v_b   = k_t  + (size_t)BB * CC * NN;      // 4194304  [b][h][d][n]
  _Float16* ao_t  = v_b  + (size_t)BB * CC * NN;      // 4194304  [b][n][c]
  (void)ws_size; (void)in_sizes; (void)n_in; (void)out_size;

  cast_weights<<<4096, 256, 0, stream>>>(qkv_w, proj_w, wq_h, wp_h);
  gn_stats    <<<BB * NG, 256, 0, stream>>>(x, stats);
  gn_apply    <<<(BB * CC * NN) / 256, 256, 0, stream>>>(x, stats, norm_w, norm_b, xn_t);
  qkv_gemm    <<<3072, 128, 0, stream>>>(wq_h, xn_t, qkv_b, q_t, k_t, v_b);
  attn_flash  <<<1024, 128, 0, stream>>>(q_t, k_t, v_b, ao_t);
  proj_gemm   <<<1024, 128, 0, stream>>>(wp_h, ao_t, proj_b, x, out);
}
